// TFM_31731218383385
// MI455X (gfx1250) — matrix operand order repaired and checked by reading, not compiled, not run on hardware
//
#include <hip/hip_runtime.h>
#include <hip/hip_bf16.h>

typedef _Float16 v16h __attribute__((ext_vector_type(16)));
typedef float    v8f  __attribute__((ext_vector_type(8)));
typedef float    v4f  __attribute__((ext_vector_type(4)));

__device__ __forceinline__ float fast_rcp(float v) { return __builtin_amdgcn_rcpf(v); }
__device__ __forceinline__ float dev_silu(float v) {
    // v * sigmoid(v) with v_exp + v_rcp (no IEEE divide chain)
    return v * fast_rcp(1.0f + __expf(-v));
}

// float atomic-max via sign-split ordered-int trick
__device__ __forceinline__ void atomicMaxF(float* addr, float val) {
    if (val >= 0.0f) atomicMax((int*)addr, __float_as_int(val));
    else             atomicMin((unsigned int*)addr, __float_as_uint(val));
}

// ---------------- elementwise / setup kernels ----------------

__global__ void k_outinit(float* out, const float* bfc) { out[0] = bfc[0]; }

// block-per-node, thread-per-feature (no div/mod)
__global__ void k_embed(const int* __restrict__ zn, const float* __restrict__ emb,
                        float* __restrict__ x, int D) {
    int n = blockIdx.x, d = threadIdx.x;
    x[(long long)n * D + d] = emb[(long long)zn[n] * D + d];
}

__global__ void k_bond(const float* __restrict__ r, float* __restrict__ bl, int E) {
    int e = blockIdx.x * blockDim.x + threadIdx.x;
    if (e >= E) return;
    float x = r[3 * e], y = r[3 * e + 1], z = r[3 * e + 2];
    bl[e] = sqrtf(x * x + y * y + z * z);
}

__global__ void k_theta(const float* __restrict__ r, const float* __restrict__ bl,
                        const int* __restrict__ ts, const int* __restrict__ td,
                        float* __restrict__ theta, int T) {
    int t = blockIdx.x * blockDim.x + threadIdx.x;
    if (t >= T) return;
    int s = ts[t], d = td[t];
    // rnorm = -r/|r| on both sides -> signs cancel
    float dot = r[3 * s] * r[3 * d] + r[3 * s + 1] * r[3 * d + 1] + r[3 * s + 2] * r[3 * d + 2];
    float c = dot * fast_rcp(bl[s] * bl[d]);
    c = fminf(1.0f, fmaxf(-1.0f, c));
    theta[t] = acosf(c);
}

// ---------------- weight pre-pack into WMMA B-fragment layout (f16) ----------------
// Packed index: (((tn*ksteps + kt)*32 + lane)*16 + h)
// element = W[(kt*32 + (lane>>4)*8 + (h&7) + (h>>3)*16)*Nc + tn*16 + (lane&15)]
// => per WMMA, each lane loads 16 contiguous halves (32 B) = two b128 loads.
__global__ void k_packW(const float* __restrict__ W, _Float16* __restrict__ Wp,
                        long long total, int ksteps, int Nc) {
    long long i = (long long)blockIdx.x * blockDim.x + threadIdx.x;
    if (i >= total) return;
    int h    = (int)(i & 15);
    int lane = (int)((i >> 4) & 31);
    long long rest = i >> 9;              // tn*ksteps + kt
    int kt = (int)(rest % ksteps);
    int tn = (int)(rest / ksteps);
    int row = lane & 15, hi = lane >> 4;
    int k = kt * 32 + hi * 8 + (h & 7) + (h >> 3) * 16;
    int n = tn * 16 + row;
    Wp[i] = (_Float16)W[(long long)k * Nc + n];
}

// ---------------- WMMA GEMM: out[M,Nc] = act(A[M,K] @ W[K,Nc] + bias) ----------------
// One wave computes a 16x(16*NT) output strip. Per K-step: A fragment loaded once,
// all NT B fragments preloaded into registers, then NT back-to-back v_wmma ops
// (single wait before the chain; next step's loads overlap the matrix pipe).
template <int ACT, int NT>
__global__ void k_gemm_wmma(const float* __restrict__ A, const _Float16* __restrict__ Wp,
                            const float* __restrict__ bias, float* __restrict__ out,
                            long long M, int K, int Nc) {
    int lane = threadIdx.x & 31;
    int wave = threadIdx.x >> 5;
    long long strip = (long long)blockIdx.x * (blockDim.x >> 5) + wave;
    int groups = (Nc >> 4) / NT;                 // column-tile groups per row strip
    long long tm = strip / groups;
    int tg = (int)(strip % groups);
    if (tm * 16 >= M) return;                    // wave-uniform (EXEC stays full)
    int row = lane & 15;
    int hi  = lane >> 4;
    const float* Ar = A + (tm * 16 + row) * (long long)K;
    int ksteps = K >> 5;
    const _Float16* Bp = Wp + (((long long)(tg * NT) * ksteps) << 9) + ((long long)lane << 4);
    long long bstride = (long long)ksteps << 9;  // halves per column tile

    v8f acc[NT];
#pragma unroll
    for (int j = 0; j < NT; ++j) acc[j] = (v8f){};

    for (int kt = 0; kt < ksteps; ++kt) {
        // ---- A fragment: 4 x b128 loads + packed f32->f16 converts ----
        int ka = kt * 32 + hi * 8;
        v4f a0 = *(const v4f*)(Ar + ka);
        v4f a1 = *(const v4f*)(Ar + ka + 4);
        v4f a2 = *(const v4f*)(Ar + ka + 16);
        v4f a3 = *(const v4f*)(Ar + ka + 20);
        v16h a;
#pragma unroll
        for (int h = 0; h < 4; ++h) {
            a[h]      = (_Float16)a0[h];
            a[h + 4]  = (_Float16)a1[h];
            a[h + 8]  = (_Float16)a2[h];
            a[h + 12] = (_Float16)a3[h];
        }
        // ---- preload ALL B fragments, then issue WMMAs back-to-back ----
        v16h bfrag[NT];
#pragma unroll
        for (int j = 0; j < NT; ++j)
            bfrag[j] = *(const v16h*)(Bp + j * bstride + ((long long)kt << 9));
#pragma unroll
        for (int j = 0; j < NT; ++j)
            acc[j] = __builtin_amdgcn_wmma_f32_16x16x32_f16(false, a, false, bfrag[j], (short)0,
                                                            acc[j], false, false);
    }
    float* orow = out + (tm * 16 + hi * 8) * (long long)Nc + tg * (NT * 16) + row;
#pragma unroll
    for (int j = 0; j < NT; ++j) {
        float bv = bias[tg * (NT * 16) + j * 16 + row];
        float* op = orow + j * 16;
#pragma unroll
        for (int q = 0; q < 8; ++q) {
            float v = acc[j][q] + bv;
            if (ACT) v = dev_silu(v);
            op[(long long)q * Nc] = v;
        }
    }
}

// RBF-fused GEMM: A[e,d] = exp(-gamma*(bl[e]-d*dc)^2) generated in-register, reused across NT tiles
template <int NT>
__global__ void k_gemm_rbf(const float* __restrict__ bl, const _Float16* __restrict__ Wp,
                           const float* __restrict__ bias, float* __restrict__ out,
                           long long M, int K, int Nc, float gamma, float dc) {
    int lane = threadIdx.x & 31;
    int wave = threadIdx.x >> 5;
    long long strip = (long long)blockIdx.x * (blockDim.x >> 5) + wave;
    int groups = (Nc >> 4) / NT;
    long long tm = strip / groups;
    int tg = (int)(strip % groups);
    if (tm * 16 >= M) return;
    int row = lane & 15;
    int hi  = lane >> 4;
    float blv = bl[tm * 16 + row];
    int ksteps = K >> 5;
    const _Float16* Bp = Wp + (((long long)(tg * NT) * ksteps) << 9) + ((long long)lane << 4);
    long long bstride = (long long)ksteps << 9;

    v8f acc[NT];
#pragma unroll
    for (int j = 0; j < NT; ++j) acc[j] = (v8f){};

    for (int kt = 0; kt < ksteps; ++kt) {
        // preload B fragments first (loads in flight while we generate A via v_exp)
        v16h bfrag[NT];
#pragma unroll
        for (int j = 0; j < NT; ++j)
            bfrag[j] = *(const v16h*)(Bp + j * bstride + ((long long)kt << 9));

        int ka = kt * 32 + hi * 8;
        v16h a;
#pragma unroll
        for (int h = 0; h < 8; ++h) {
            float d0 = blv - dc * (float)(ka + h);
            float d1 = blv - dc * (float)(ka + 16 + h);
            a[h]     = (_Float16)__expf(-gamma * d0 * d0);
            a[h + 8] = (_Float16)__expf(-gamma * d1 * d1);
        }
#pragma unroll
        for (int j = 0; j < NT; ++j)
            acc[j] = __builtin_amdgcn_wmma_f32_16x16x32_f16(false, a, false, bfrag[j], (short)0,
                                                            acc[j], false, false);
    }
    float* orow = out + (tm * 16 + hi * 8) * (long long)Nc + tg * (NT * 16) + row;
#pragma unroll
    for (int j = 0; j < NT; ++j) {
        float bv = bias[tg * (NT * 16) + j * 16 + row];
        float* op = orow + j * 16;
#pragma unroll
        for (int q = 0; q < 8; ++q)
            op[(long long)q * Nc] = acc[j][q] + bv;
    }
}

// ---------------- edge / triplet kernels ----------------

// block-per-edge, thread-per-channel
__global__ void k_edge(float* __restrict__ xij, const float* __restrict__ xj,
                       const float* __restrict__ xi, const int* __restrict__ gs,
                       const int* __restrict__ gd, int DM) {
    int e = blockIdx.x, k = threadIdx.x;
    long long i = (long long)e * DM + k;
    xij[i] += xj[(long long)gs[e] * DM + k] + xi[(long long)gd[e] * DM + k];
}

__global__ void k_linit(float* __restrict__ m, float* __restrict__ den,
                        float* __restrict__ ft, int E, long long ftN) {
    long long i = (long long)blockIdx.x * blockDim.x + threadIdx.x;
    if (i < E) { m[i] = -3.0e38f; den[i] = 0.0f; }
    if (i < ftN) ft[i] = 0.0f;
}

// wave-per-triplet: z_k = cos(k*theta) regenerated, silu, dot with attn, wave-reduce
__global__ void k_logit(const float* __restrict__ theta, const float* __restrict__ xij,
                        const int* __restrict__ ts, const int* __restrict__ td,
                        const float* __restrict__ attn, float* __restrict__ logit,
                        float* __restrict__ m, int T, int DM) {
    long long gt = (long long)blockIdx.x * blockDim.x + threadIdx.x;
    long long t = gt >> 5;
    int lane = threadIdx.x & 31;
    if (t >= T) return;
    int s = ts[t], d = td[t];
    float th = theta[t];
    const float* xs = xij + (long long)s * DM;
    const float* xd = xij + (long long)d * DM;
    float p = 0.0f;
    for (int k = lane; k < DM; k += 32) {
        float e = __cosf((float)k * th) + xs[k] + xd[k];
        p += dev_silu(e) * attn[k];
    }
#pragma unroll
    for (int off = 16; off; off >>= 1) p += __shfl_xor(p, off, 32);
    if (lane == 0) { logit[t] = p; atomicMaxF(&m[d], p); }
}

__global__ void k_expnorm(float* __restrict__ logit, const float* __restrict__ m,
                          float* __restrict__ den, const int* __restrict__ td, int T) {
    int t = blockIdx.x * blockDim.x + threadIdx.x;
    if (t >= T) return;
    int d = td[t];
    float ex = __expf(logit[t] - m[d]);
    logit[t] = ex;
    atomicAdd(&den[d], ex);
}

__global__ void k_scatter(const float* __restrict__ ex, const float* __restrict__ den,
                          const float* __restrict__ xij, const int* __restrict__ ts,
                          const int* __restrict__ td, const int* __restrict__ gd,
                          float* __restrict__ ft, int T, int DM) {
    long long gt = (long long)blockIdx.x * blockDim.x + threadIdx.x;
    long long t = gt >> 5;
    int lane = threadIdx.x & 31;
    if (t >= T) return;
    int s = ts[t], d = td[t];
    int node = gd[d];
    float a = ex[t] * fast_rcp(den[d]);
    const float* xs = xij + (long long)s * DM;
    float* fo = ft + (long long)node * DM;
    for (int k = lane; k < DM; k += 32)
        atomicAdd(&fo[k], a * xs[k]);
}

// wave-per-node readout: out += sum_n (x[n]·Wfc) * (1/N)   (bfc pre-seeded)
__global__ void k_energy(const float* __restrict__ x, const float* __restrict__ Wfc,
                         float* __restrict__ out, int N, int D, float invN) {
    long long gt = (long long)blockIdx.x * blockDim.x + threadIdx.x;
    long long n = gt >> 5;
    int lane = threadIdx.x & 31;
    if (n >= N) return;
    const float* xr = x + n * (long long)D;
    float p = 0.0f;
    for (int k = lane; k < D; k += 32) p += xr[k] * Wfc[k];
#pragma unroll
    for (int off = 16; off; off >>= 1) p += __shfl_xor(p, off, 32);
    if (lane == 0) atomicAdd(out, p * invN);
}

// ---------------- host launcher ----------------

extern "C" void kernel_launch(void* const* d_in, const int* in_sizes, int n_in,
                              void* d_out, int out_size, void* d_ws, size_t ws_size,
                              hipStream_t stream) {
    const int*   zn    = (const int*)  d_in[0];
    const int*   g_src = (const int*)  d_in[1];
    const int*   g_dst = (const int*)  d_in[2];
    const int*   t_src = (const int*)  d_in[3];
    const int*   t_dst = (const int*)  d_in[4];
    const float* r     = (const float*)d_in[5];
    const float* emb   = (const float*)d_in[6];
    const float* Wsrc  = (const float*)d_in[7];
    const float* bsrc  = (const float*)d_in[8];
    const float* Wdst  = (const float*)d_in[9];
    const float* bdst  = (const float*)d_in[10];
    const float* Wedge = (const float*)d_in[11];
    const float* bedge = (const float*)d_in[12];
    const float* attn  = (const float*)d_in[13];
    const float* W1    = (const float*)d_in[14];
    const float* b1    = (const float*)d_in[15];
    const float* W2    = (const float*)d_in[16];
    const float* b2    = (const float*)d_in[17];
    const float* Wfc   = (const float*)d_in[18];
    const float* bfc   = (const float*)d_in[19];

    const int N  = in_sizes[0];
    const int E  = in_sizes[1];
    const int T  = in_sizes[3];
    const int D  = in_sizes[18];         // Wfc is [D,1]
    const int L  = in_sizes[17] / D;     // b2   is [L,D]
    const int DM = in_sizes[8]  / L;     // bsrc is [L,DM]
    const int H  = in_sizes[15] / L;     // b1   is [L,H]

    // scratch carve-out
    char* base = (char*)d_ws;
    size_t off = 0;
    auto allocB = [&](size_t bytes) -> void* {
        off = (off + 255) & ~(size_t)255;
        void* p = (void*)(base + off);
        off += bytes;
        return p;
    };
    auto allocF = [&](size_t elems) -> float* { return (float*)allocB(elems * sizeof(float)); };

    float* x     = allocF((size_t)N * D);
    float* bl    = allocF((size_t)E);
    float* theta = allocF((size_t)T);
    float* xj    = allocF((size_t)N * DM);
    float* xi    = allocF((size_t)N * DM);
    float* xij   = allocF((size_t)E * DM);   // rbf-GEMM output, updated in place
    float* logit = allocF((size_t)T);
    float* mseg  = allocF((size_t)E);
    float* den   = allocF((size_t)E);
    float* ft    = allocF((size_t)N * DM);
    float* hbuf  = allocF((size_t)N * H);
    size_t maxKN = (size_t)D * DM;
    if ((size_t)DM * H > maxKN) maxKN = (size_t)DM * H;
    if ((size_t)H * D  > maxKN) maxKN = (size_t)H * D;
    _Float16* wpack = (_Float16*)allocB(maxKN * sizeof(_Float16));
    (void)ws_size; (void)n_in; (void)out_size;

    const int TB = 256;
    auto nblk = [](long long n, int tb) { return (unsigned)((n + tb - 1) / tb); };

    k_outinit<<<1, 1, 0, stream>>>((float*)d_out, bfc);
    k_embed<<<N, D, 0, stream>>>(zn, emb, x, D);
    k_bond<<<nblk(E, TB), TB, 0, stream>>>(r, bl, E);
    k_theta<<<nblk(T, TB), TB, 0, stream>>>(r, bl, t_src, t_dst, theta, T);

    auto pack = [&](const float* W, int K, int Nc) {
        long long total = (long long)K * Nc;
        k_packW<<<nblk(total, TB), TB, 0, stream>>>(W, wpack, total, K >> 5, Nc);
    };
    auto gemm = [&](const float* A, const float* bias, float* out,
                    long long M, int K, int Nc, int act) {
        int tilesN = Nc >> 4;
        if ((tilesN & 3) == 0) {
            long long strips = (M / 16) * (tilesN >> 2);
            unsigned nb = (unsigned)((strips + 3) / 4);     // 4 waves / block
            if (act) k_gemm_wmma<1, 4><<<nb, 128, 0, stream>>>(A, wpack, bias, out, M, K, Nc);
            else     k_gemm_wmma<0, 4><<<nb, 128, 0, stream>>>(A, wpack, bias, out, M, K, Nc);
        } else {
            long long strips = (M / 16) * tilesN;
            unsigned nb = (unsigned)((strips + 3) / 4);
            if (act) k_gemm_wmma<1, 1><<<nb, 128, 0, stream>>>(A, wpack, bias, out, M, K, Nc);
            else     k_gemm_wmma<0, 1><<<nb, 128, 0, stream>>>(A, wpack, bias, out, M, K, Nc);
        }
    };

    const float dc    = 8.0f / (float)(D - 1);
    const float gamma = 1.0f / (dc * dc);

    for (int l = 0; l < L; ++l) {
        pack(Wsrc + (size_t)l * D * DM, D, DM);
        gemm(x, bsrc + (size_t)l * DM, xj, N, D, DM, 0);
        pack(Wdst + (size_t)l * D * DM, D, DM);
        gemm(x, bdst + (size_t)l * DM, xi, N, D, DM, 0);

        pack(Wedge + (size_t)l * D * DM, D, DM);
        {
            int tilesN = DM >> 4;
            if ((tilesN & 3) == 0) {
                long long strips = ((long long)E / 16) * (tilesN >> 2);
                unsigned nb = (unsigned)((strips + 3) / 4);
                k_gemm_rbf<4><<<nb, 128, 0, stream>>>(bl, wpack, bedge + (size_t)l * DM, xij,
                                                      (long long)E, D, DM, gamma, dc);
            } else {
                long long strips = ((long long)E / 16) * tilesN;
                unsigned nb = (unsigned)((strips + 3) / 4);
                k_gemm_rbf<1><<<nb, 128, 0, stream>>>(bl, wpack, bedge + (size_t)l * DM, xij,
                                                      (long long)E, D, DM, gamma, dc);
            }
        }
        k_edge<<<E, DM, 0, stream>>>(xij, xj, xi, g_src, g_dst, DM);

        long long ftN = (long long)N * DM;
        long long initN = ftN > E ? ftN : E;
        k_linit<<<nblk(initN, TB), TB, 0, stream>>>(mseg, den, ft, E, ftN);

        k_logit<<<nblk((long long)T * 32, TB), TB, 0, stream>>>(
            theta, xij, t_src, t_dst, attn + (size_t)l * DM, logit, mseg, T, DM);
        k_expnorm<<<nblk(T, TB), TB, 0, stream>>>(logit, mseg, den, t_dst, T);
        k_scatter<<<nblk((long long)T * 32, TB), TB, 0, stream>>>(
            logit, den, xij, t_src, t_dst, g_dst, ft, T, DM);

        pack(W1 + (size_t)l * DM * H, DM, H);
        gemm(ft, b1 + (size_t)l * H, hbuf, N, DM, H, 1);
        pack(W2 + (size_t)l * H * D, H, D);
        gemm(hbuf, b2 + (size_t)l * D, x, N, H, D, 0);
    }

    k_energy<<<nblk((long long)N * 32, TB), TB, 0, stream>>>(x, Wfc, (float*)d_out, N, D,
                                                             1.0f / (float)N);
}
